// CMAModel_6777458393277
// MI455X (gfx1250) — matrix operand order repaired and checked by reading, not compiled, not run on hardware
//
#include <hip/hip_runtime.h>

// ---------------- problem constants ----------------
#define HN    16
#define CDIM  1024
#define DH    64
#define BN    2
#define TN    2048
#define MMEM  512
#define SN    (TN + 2*MMEM)   // 3072

typedef __attribute__((ext_vector_type(16))) _Float16 v16h;
typedef __attribute__((ext_vector_type(8)))  _Float16 v8h;
typedef __attribute__((ext_vector_type(8)))  float    v8f;
typedef __attribute__((ext_vector_type(4))) unsigned int u32x4;
typedef __attribute__((ext_vector_type(8))) int          i32x8;
typedef __attribute__((ext_vector_type(4))) int          i32x4;

// ---------------- WMMA fragment helpers (wave32) ----------------
// A-matrix 16x32 f16 layout (ISA 7.12.2): lane<16 -> row=lane, K=0..7 then 16..23;
// lane>=16 -> row=lane-16, K=8..15 then 24..31.
__device__ __forceinline__ v16h load_fragA(const _Float16* base, int ld, int k0) {
  int lane = threadIdx.x & 31;
  int r    = lane & 15;
  int half = lane >> 4;
  const _Float16* p = base + (size_t)r * ld + k0 + 8 * half;
  v8h lo = *(const v8h*)(p);
  v8h hi = *(const v8h*)(p + 16);
  return __builtin_shufflevector(lo, hi, 0,1,2,3,4,5,6,7,8,9,10,11,12,13,14,15);
}

// B-matrix 32x16 f16: column n = lane&15; lane<16 holds K=0..7 then 16..23, lane>=16 holds K=8..15 then 24..31.
// `base` points at the row-major matrix whose row n equals B's column n.
__device__ __forceinline__ v16h load_fragB(const _Float16* base, int ld, int k0) {
  int lane = threadIdx.x & 31;
  int r    = lane & 15;
  int half = lane >> 4;
  const _Float16* p = base + (size_t)r * ld + k0 + 8 * half;
  v8h lo = *(const v8h*)(p);
  v8h hi = *(const v8h*)(p + 16);
  return __builtin_shufflevector(lo, hi, 0,1,2,3,4,5,6,7,8,9,10,11,12,13,14,15);
}

__device__ __forceinline__ v8f wmma32(v16h a, v16h b, v8f c) {
  return __builtin_amdgcn_wmma_f32_16x16x32_f16(false, a, false, b, (short)0, c, false, false);
}

// ---------------- TDM: DMA a 32x64 f16 tile (row stride CDIM) from global to LDS ----
// D# per CDNA5 ISA ch.8: group0 = {flags, lds_addr, global_addr, type=2},
// group1 = {data_size=2B, tensor_dim0/1, tile_dim0/1, tensor_dim0_stride}.
// 6-arg builtin form (clang-23 / therock headers).
__device__ __forceinline__ void tdm_load_tile_32x64(unsigned ldsoff, const _Float16* gsrc) {
  unsigned long long ga = (unsigned long long)(uintptr_t)gsrc;
  u32x4 g0;
  g0[0] = 1u;                                                  // count=1, user mode
  g0[1] = ldsoff;                                              // lds_addr (bytes)
  g0[2] = (unsigned)ga;                                        // global_addr[31:0]
  g0[3] = (unsigned)((ga >> 32) & 0x01FFFFFFu) | (2u << 30);   // addr[56:32] | type=2
  i32x8 g1;
  g1[0] = 0x10000;        // workgroup_mask=0, data_size=1 (2 bytes)
  g1[1] = (64 << 16);     // tensor_dim0[15:0] = 64
  g1[2] = (32 << 16);     // tensor_dim0 hi = 0 | tensor_dim1[15:0] = 32
  g1[3] = (64 << 16);     // tensor_dim1 hi = 0 | tile_dim0 = 64
  g1[4] = 32;             // tile_dim1 = 32, tile_dim2 = 0
  g1[5] = CDIM;           // tensor_dim0_stride[31:0] = 1024 elements
  g1[6] = 0;
  g1[7] = 0;
  i32x4 z4 = {0, 0, 0, 0};
  i32x8 z8 = {0, 0, 0, 0, 0, 0, 0, 0};
  __builtin_amdgcn_tensor_load_to_lds(g0, g1, z4, z4, z8, 0);
}

// ---------------- conversion kernels ----------------
__global__ void cvt_kernel(const float* __restrict__ in, _Float16* __restrict__ out, int n) {
  int i = blockIdx.x * 256 + threadIdx.x;
  if (i < n) out[i] = (_Float16)in[i];
}

__global__ void build_ksrc(const float* __restrict__ x, const float* __restrict__ fm,
                           const float* __restrict__ rm,
                           _Float16* __restrict__ ksrc, _Float16* __restrict__ xh) {
  size_t idx = (size_t)blockIdx.x * 256 + threadIdx.x;
  if (idx >= (size_t)BN * SN * CDIM) return;
  int c = idx % CDIM;
  size_t bs = idx / CDIM;
  int s = bs % SN;
  int b = bs / SN;
  float v;
  if (s < TN) {
    v = x[((size_t)b * TN + s) * CDIM + c];
    xh[((size_t)b * TN + s) * CDIM + c] = (_Float16)v;
  } else if (s < TN + MMEM) {
    v = fm[((size_t)b * MMEM + (s - TN)) * CDIM + c];
  } else {
    v = rm[((size_t)b * MMEM + (s - TN - MMEM)) * CDIM + c];
  }
  ksrc[idx] = (_Float16)v;
}

__global__ void zero1(float* p) { if (threadIdx.x == 0 && blockIdx.x == 0) *p = 0.f; }

// ---------------- WMMA GEMM: C[M,N] = A[M,K] * W[N,K]^T ----------------
// Block: 256 threads (8 waves), tile M=128 x N=64. W tile staged through LDS and
// shared by all 8 waves. M and N must be multiples of 128 / 64 (true here).
template<bool F32OUT>
__global__ void gemm16(const _Float16* __restrict__ A, const _Float16* __restrict__ W,
                       float* __restrict__ Co, _Float16* __restrict__ Ch,
                       int Mrows, int Ncols, int Kdim) {
  __shared__ __align__(32) _Float16 Wt[64][32];
  int wave = threadIdx.x >> 5;
  int m0 = blockIdx.y * 128 + wave * 16;
  int n0 = blockIdx.x * 64;
  v8f acc[4] = {};
  const _Float16* Abase = A + (size_t)m0 * Kdim;
  int wr   = threadIdx.x >> 2;        // 0..63 : W row within tile
  int woff = (threadIdx.x & 3) * 8;   // 0,8,16,24
  const _Float16* wsrc = W + (size_t)(n0 + wr) * Kdim + woff;
  for (int k0 = 0; k0 < Kdim; k0 += 32) {
    __syncthreads();
    *(v8h*)(&Wt[wr][woff]) = *(const v8h*)(wsrc + k0);
    if (k0 + 32 < Kdim) __builtin_prefetch(wsrc + k0 + 32, 0, 0);
    __syncthreads();
    v16h a  = load_fragA(Abase, Kdim, k0);
    v16h b0 = load_fragB(&Wt[0][0],  32, 0);
    v16h b1 = load_fragB(&Wt[16][0], 32, 0);
    v16h b2 = load_fragB(&Wt[32][0], 32, 0);
    v16h b3 = load_fragB(&Wt[48][0], 32, 0);
    acc[0] = wmma32(a, b0, acc[0]);
    acc[1] = wmma32(a, b1, acc[1]);
    acc[2] = wmma32(a, b2, acc[2]);
    acc[3] = wmma32(a, b3, acc[3]);
  }
  int lane = threadIdx.x & 31;
  int col  = lane & 15;
  int row0 = (lane >> 4) * 8;
#pragma unroll
  for (int j = 0; j < 4; ++j)
#pragma unroll
    for (int i = 0; i < 8; ++i) {
      size_t idx = (size_t)(m0 + row0 + i) * Ncols + n0 + 16 * j + col;
      if (F32OUT) Co[idx] = acc[j][i];
      else        Ch[idx] = (_Float16)acc[j][i];
    }
  (void)Mrows;
}

// ---------------- gate: g = sigmoid(q @ gate_W^T + b), loss = 0.01*mean(g) ----------------
__global__ void gate_kernel(const _Float16* __restrict__ qh, const _Float16* __restrict__ gw,
                            const float* __restrict__ gate_b, float* __restrict__ gbuf,
                            float* __restrict__ loss) {
  int idx = blockIdx.x * 256 + threadIdx.x;     // over B*T*H = 65536
  int h  = idx & (HN - 1);
  int bt = idx >> 4;
  const v8h* qrow = (const v8h*)(qh + (size_t)bt * CDIM);
  const v8h* wrow = (const v8h*)(gw + (size_t)h * CDIM);
  float acc = 0.f;
  for (int c = 0; c < CDIM / 8; ++c) {
    v8h a = qrow[c], w = wrow[c];
#pragma unroll
    for (int j = 0; j < 8; ++j) acc += (float)a[j] * (float)w[j];
  }
  float g = 1.f / (1.f + __expf(-(acc + gate_b[h])));
  gbuf[idx] = g;
  __shared__ float red[256];
  red[threadIdx.x] = g;
  __syncthreads();
  for (int s = 128; s > 0; s >>= 1) {
    if (threadIdx.x < s) red[threadIdx.x] += red[threadIdx.x + s];
    __syncthreads();
  }
  if (threadIdx.x == 0) atomicAdd(loss, red[0] * (0.01f / ((float)BN * TN * HN)));
}

// ---------------- flash attention with memory gate ----------------
// 128 threads = 4 waves; wave w owns t-rows [blockIdx.x*64 + 16w, +16).
// K tile is DMA'd to LDS by the Tensor Data Mover (wave 0 issues, s_wait_tensorcnt).
// Phase 1: causal chunk keys (s < T)  -> accC only.
// Phase 2: memory keys (s >= T)       -> accM, rescaling both.
__global__ void attention_kernel(const _Float16* __restrict__ qh,
                                 const _Float16* __restrict__ kh,
                                 const _Float16* __restrict__ vh,
                                 const float* __restrict__ gbuf,
                                 float* __restrict__ attn) {
  __shared__ __align__(32) _Float16 Kt[32][64];      // K tile, row-major (s_local, d)
  __shared__ __align__(32) _Float16 Vt[64][32];      // V tile, transposed (d, s_local)
  __shared__ __align__(32) _Float16 Pb[4][16][32];   // per-wave P staging

  int b = blockIdx.z, h = blockIdx.y;
  int wave = threadIdx.x >> 5;
  int lane = threadIdx.x & 31;
  int t0w = blockIdx.x * 64 + wave * 16;
  const float sl2 = 0.125f * 1.44269504f;            // (1/sqrt(64)) * log2(e)
  unsigned kt_off = (unsigned)(size_t)&Kt[0][0];     // LDS byte offset of K tile

  const _Float16* qbase = qh + ((size_t)(b * TN + t0w)) * CDIM + h * DH;
  v16h qa0 = load_fragA(qbase, CDIM, 0);
  v16h qa1 = load_fragA(qbase, CDIM, 32);

  v8f accC[4] = {};
  float rmax[8], rsum[8];
#pragma unroll
  for (int i = 0; i < 8; ++i) { rmax[i] = -1e30f; rsum[i] = 0.f; }

  const _Float16* kbh = kh + ((size_t)b * SN) * CDIM + h * DH;
  const _Float16* vbh = vh + ((size_t)b * SN) * CDIM + h * DH;
  int colw = lane & 15, rw = (lane >> 4) * 8;

  // ================= phase 1: causal region =================
  int chunk_end = blockIdx.x * 64 + 64;   // multiple of 32, <= TN
  for (int s0 = 0; s0 < chunk_end; s0 += 32) {
    __syncthreads();
    if (threadIdx.x < 32)                              // wave 0 owns the TDM issue
      tdm_load_tile_32x64(kt_off, kbh + (size_t)s0 * CDIM);
#pragma unroll
    for (int it = 0; it < 2; ++it) {                   // V tile, transposed store
      int chunk = threadIdx.x + it * 128;
      int sl = chunk & 31, d0 = (chunk >> 5) * 8;
      v8h tmp = *(const v8h*)(vbh + (size_t)(s0 + sl) * CDIM + d0);
      __builtin_prefetch(vbh + (size_t)(s0 + 32 + sl) * CDIM + d0, 0, 0);
#pragma unroll
      for (int q = 0; q < 8; ++q) Vt[d0 + q][sl] = tmp[q];
    }
    if (threadIdx.x < 32) __builtin_amdgcn_s_wait_tensorcnt(0);
    __syncthreads();

    if (s0 < t0w + 16) {                 // wave-uniform: this wave needs the chunk
      v8f sc[2];
#pragma unroll
      for (int sh = 0; sh < 2; ++sh) {
        const _Float16* kb = &Kt[sh * 16][0];
        v16h kb0 = load_fragB(kb, 64, 0);
        v16h kb1 = load_fragB(kb, 64, 32);
        v8f s_ = {};
        s_ = wmma32(qa0, kb0, s_);
        s_ = wmma32(qa1, kb1, s_);
#pragma unroll
        for (int i = 0; i < 8; ++i) s_[i] *= sl2;
        if (s0 + 31 > t0w) {             // diagonal chunk: apply causal mask
          int dcol = (s0 + sh * 16 + colw) - (t0w + rw);
#pragma unroll
          for (int i = 0; i < 8; ++i)
            if (dcol > i) s_[i] = -1e30f;
        }
        sc[sh] = s_;
      }
      float m8[8];
#pragma unroll
      for (int i = 0; i < 8; ++i) m8[i] = fmaxf(sc[0][i], sc[1][i]);
#pragma unroll
      for (int msk = 1; msk < 16; msk <<= 1)
#pragma unroll
        for (int i = 0; i < 8; ++i) m8[i] = fmaxf(m8[i], __shfl_xor(m8[i], msk, 32));
#pragma unroll
      for (int i = 0; i < 8; ++i) {
        float mnew = fmaxf(rmax[i], m8[i]);
        float corr = exp2f(rmax[i] - mnew);
        rmax[i] = mnew;
        sc[0][i] = exp2f(sc[0][i] - mnew);
        sc[1][i] = exp2f(sc[1][i] - mnew);
        rsum[i] *= corr;
#pragma unroll
        for (int j = 0; j < 4; ++j) accC[j][i] *= corr;
        m8[i] = sc[0][i] + sc[1][i];
      }
#pragma unroll
      for (int msk = 1; msk < 16; msk <<= 1)
#pragma unroll
        for (int i = 0; i < 8; ++i) m8[i] += __shfl_xor(m8[i], msk, 32);
#pragma unroll
      for (int i = 0; i < 8; ++i) rsum[i] += m8[i];

#pragma unroll
      for (int sh = 0; sh < 2; ++sh)
#pragma unroll
        for (int i = 0; i < 8; ++i)
          Pb[wave][rw + i][sh * 16 + colw] = (_Float16)sc[sh][i];
      v16h pa  = load_fragA(&Pb[wave][0][0], 32, 0);
      v16h vb0 = load_fragB(&Vt[0][0],  32, 0);
      v16h vb1 = load_fragB(&Vt[16][0], 32, 0);
      v16h vb2 = load_fragB(&Vt[32][0], 32, 0);
      v16h vb3 = load_fragB(&Vt[48][0], 32, 0);
      accC[0] = wmma32(pa, vb0, accC[0]);
      accC[1] = wmma32(pa, vb1, accC[1]);
      accC[2] = wmma32(pa, vb2, accC[2]);
      accC[3] = wmma32(pa, vb3, accC[3]);
    }
  }

  // ================= phase 2: memory region (no mask) =================
  v8f accM[4] = {};
  for (int s0 = TN; s0 < SN; s0 += 32) {
    __syncthreads();
    if (threadIdx.x < 32)
      tdm_load_tile_32x64(kt_off, kbh + (size_t)s0 * CDIM);
#pragma unroll
    for (int it = 0; it < 2; ++it) {
      int chunk = threadIdx.x + it * 128;
      int sl = chunk & 31, d0 = (chunk >> 5) * 8;
      v8h tmp = *(const v8h*)(vbh + (size_t)(s0 + sl) * CDIM + d0);
      if (s0 + 32 < SN)
        __builtin_prefetch(vbh + (size_t)(s0 + 32 + sl) * CDIM + d0, 0, 0);
#pragma unroll
      for (int q = 0; q < 8; ++q) Vt[d0 + q][sl] = tmp[q];
    }
    if (threadIdx.x < 32) __builtin_amdgcn_s_wait_tensorcnt(0);
    __syncthreads();

    v8f sc[2];
#pragma unroll
    for (int sh = 0; sh < 2; ++sh) {
      const _Float16* kb = &Kt[sh * 16][0];
      v16h kb0 = load_fragB(kb, 64, 0);
      v16h kb1 = load_fragB(kb, 64, 32);
      v8f s_ = {};
      s_ = wmma32(qa0, kb0, s_);
      s_ = wmma32(qa1, kb1, s_);
#pragma unroll
      for (int i = 0; i < 8; ++i) s_[i] *= sl2;
      sc[sh] = s_;
    }
    float m8[8];
#pragma unroll
    for (int i = 0; i < 8; ++i) m8[i] = fmaxf(sc[0][i], sc[1][i]);
#pragma unroll
    for (int msk = 1; msk < 16; msk <<= 1)
#pragma unroll
      for (int i = 0; i < 8; ++i) m8[i] = fmaxf(m8[i], __shfl_xor(m8[i], msk, 32));
#pragma unroll
    for (int i = 0; i < 8; ++i) {
      float mnew = fmaxf(rmax[i], m8[i]);
      float corr = exp2f(rmax[i] - mnew);
      rmax[i] = mnew;
      sc[0][i] = exp2f(sc[0][i] - mnew);
      sc[1][i] = exp2f(sc[1][i] - mnew);
      rsum[i] *= corr;
#pragma unroll
      for (int j = 0; j < 4; ++j) { accC[j][i] *= corr; accM[j][i] *= corr; }
      m8[i] = sc[0][i] + sc[1][i];
    }
#pragma unroll
    for (int msk = 1; msk < 16; msk <<= 1)
#pragma unroll
      for (int i = 0; i < 8; ++i) m8[i] += __shfl_xor(m8[i], msk, 32);
#pragma unroll
    for (int i = 0; i < 8; ++i) rsum[i] += m8[i];

#pragma unroll
    for (int sh = 0; sh < 2; ++sh)
#pragma unroll
      for (int i = 0; i < 8; ++i)
        Pb[wave][rw + i][sh * 16 + colw] = (_Float16)sc[sh][i];
    v16h pa  = load_fragA(&Pb[wave][0][0], 32, 0);
    v16h vb0 = load_fragB(&Vt[0][0],  32, 0);
    v16h vb1 = load_fragB(&Vt[16][0], 32, 0);
    v16h vb2 = load_fragB(&Vt[32][0], 32, 0);
    v16h vb3 = load_fragB(&Vt[48][0], 32, 0);
    accM[0] = wmma32(pa, vb0, accM[0]);
    accM[1] = wmma32(pa, vb1, accM[1]);
    accM[2] = wmma32(pa, vb2, accM[2]);
    accM[3] = wmma32(pa, vb3, accM[3]);
  }

  // epilogue: Y = (Y_chunk + g * Y_mem) / denom
  float gv[8], inv[8];
#pragma unroll
  for (int i = 0; i < 8; ++i) {
    gv[i]  = gbuf[((size_t)(b * TN + t0w + rw + i)) * HN + h];
    inv[i] = 1.f / rsum[i];
  }
#pragma unroll
  for (int j = 0; j < 4; ++j)
#pragma unroll
    for (int i = 0; i < 8; ++i) {
      float y = (accC[j][i] + gv[i] * accM[j][i]) * inv[i];
      attn[((size_t)(b * TN + t0w + rw + i)) * CDIM + h * DH + j * 16 + colw] = y;
    }
}

// ---------------- causal depthwise conv residual, emit f16 for final GEMM ----------------
__global__ void conv_kernel(const float* __restrict__ attn, const float* __restrict__ cw,
                            const float* __restrict__ cb, _Float16* __restrict__ out) {
  size_t idx = (size_t)blockIdx.x * 256 + threadIdx.x;
  if (idx >= (size_t)BN * TN * CDIM) return;
  int c = idx % CDIM;
  size_t bt = idx / CDIM;
  int t = bt % TN;
  float s = cb[c];
#pragma unroll
  for (int j = 0; j < 4; ++j) {
    int tt = t - 3 + j;
    if (tt >= 0) s += cw[c * 4 + j] * attn[idx + (size_t)(j - 3) * CDIM];
  }
  out[idx] = (_Float16)(attn[idx] + s);
}

// ---------------- launcher ----------------
extern "C" void kernel_launch(void* const* d_in, const int* in_sizes, int n_in,
                              void* d_out, int out_size, void* d_ws, size_t ws_size,
                              hipStream_t stream) {
  (void)in_sizes; (void)n_in; (void)out_size; (void)ws_size;
  const float* x  = (const float*)d_in[0];
  const float* fm = (const float*)d_in[1];
  const float* rm = (const float*)d_in[2];
  const float* Wq = (const float*)d_in[3];
  const float* Wk = (const float*)d_in[4];
  const float* Wv = (const float*)d_in[5];
  const float* Wo = (const float*)d_in[6];
  const float* gW = (const float*)d_in[7];
  const float* gb = (const float*)d_in[8];
  const float* cw = (const float*)d_in[9];
  const float* cb = (const float*)d_in[10];
  float* out = (float*)d_out;

  const size_t nBSC = (size_t)BN * SN * CDIM;   // 6291456
  const size_t nBTC = (size_t)BN * TN * CDIM;   // 4194304
  const size_t nW   = (size_t)CDIM * CDIM;      // 1048576

  char* p = (char*)d_ws;
  auto alloc = [&](size_t bytes) { void* r = (void*)p; p += (bytes + 255) & ~(size_t)255; return r; };
  _Float16* ksrc_h = (_Float16*)alloc(nBSC * 2);
  _Float16* x_h    = (_Float16*)alloc(nBTC * 2);
  _Float16* wq_h   = (_Float16*)alloc(nW * 2);
  _Float16* wk_h   = (_Float16*)alloc(nW * 2);
  _Float16* wv_h   = (_Float16*)alloc(nW * 2);
  _Float16* wo_h   = (_Float16*)alloc(nW * 2);
  _Float16* gw_h   = (_Float16*)alloc((size_t)HN * CDIM * 2);
  _Float16* q_h    = (_Float16*)alloc(nBTC * 2);
  _Float16* k_h    = (_Float16*)alloc(nBSC * 2);
  _Float16* v_h    = (_Float16*)alloc(nBSC * 2);
  float*    gbuf   = (float*)alloc((size_t)BN * TN * HN * 4);
  float*    attnf  = (float*)alloc(nBTC * 4);
  _Float16* attn2h = (_Float16*)alloc(nBTC * 2);

  build_ksrc<<<(unsigned)((nBSC + 255) / 256), 256, 0, stream>>>(x, fm, rm, ksrc_h, x_h);
  cvt_kernel<<<(unsigned)((nW + 255) / 256), 256, 0, stream>>>(Wq, wq_h, (int)nW);
  cvt_kernel<<<(unsigned)((nW + 255) / 256), 256, 0, stream>>>(Wk, wk_h, (int)nW);
  cvt_kernel<<<(unsigned)((nW + 255) / 256), 256, 0, stream>>>(Wv, wv_h, (int)nW);
  cvt_kernel<<<(unsigned)((nW + 255) / 256), 256, 0, stream>>>(Wo, wo_h, (int)nW);
  cvt_kernel<<<(HN * CDIM + 255) / 256, 256, 0, stream>>>(gW, gw_h, HN * CDIM);

  gemm16<false><<<dim3(16, 32), 256, 0, stream>>>(x_h,    wq_h, nullptr, q_h, BN * TN, CDIM, CDIM);
  gemm16<false><<<dim3(16, 48), 256, 0, stream>>>(ksrc_h, wk_h, nullptr, k_h, BN * SN, CDIM, CDIM);
  gemm16<false><<<dim3(16, 48), 256, 0, stream>>>(ksrc_h, wv_h, nullptr, v_h, BN * SN, CDIM, CDIM);

  zero1<<<1, 1, 0, stream>>>(out + nBTC);
  gate_kernel<<<(BN * TN * HN) / 256, 256, 0, stream>>>(q_h, gw_h, gb, gbuf, out + nBTC);

  attention_kernel<<<dim3(TN / 64, HN, BN), 128, 0, stream>>>(q_h, k_h, v_h, gbuf, attnf);

  conv_kernel<<<(unsigned)((nBTC + 255) / 256), 256, 0, stream>>>(attnf, cw, cb, attn2h);
  gemm16<true><<<dim3(16, 32), 256, 0, stream>>>(attn2h, wo_h, out, nullptr, BN * TN, CDIM, CDIM);
}
